// MultiheadAttention_63788854280318
// MI455X (gfx1250) — compile-verified, matrix operand order checked by reading, not run on hardware
//
#include <hip/hip_runtime.h>
#include <hip/hip_bf16.h>

// ---------------------------------------------------------------------------
// MHA forward for MI455X (gfx1250, wave32, WMMA).
// Bandwidth-bound on the 536MB attn_bias stream (~27us floor @ 23.3 TB/s).
// Matmuls: bf16 WMMA (v_wmma_f32_16x16x32_bf16), f32 accumulate.
// Attention: 64-wide S chunks, K/V tiles staged per-block into LDS with
// double-buffered global_load_async_to_lds_b128 (ASYNCcnt) to overlap the
// DMA with softmax VALU and kill the 8x redundant per-wave K/V loads.
// ---------------------------------------------------------------------------

#define DIMN 1024
#define NH   16
#define HD   64
#define BB   2
#define LL   2048
#define SS   2048
#define SCALE_F 0.125f  // 64^-0.5

typedef __attribute__((ext_vector_type(16))) __bf16 v16bf;
typedef __attribute__((ext_vector_type(8)))  float  v8f;
typedef __attribute__((ext_vector_type(4)))  unsigned int u32x4;

// ---- fragment loaders (ISA 7.12.2: 16-bit A-matrix 16x32 layout) ----------
// lane l: row m = l&15; k-half kh = l>>4; elements cover k = kh*8..kh*8+7 and
// k = 16+kh*8..16+kh*8+7 -> two 16-byte contiguous chunks per lane.
__device__ __forceinline__ v16bf load_a_bf16(const __bf16* p, int stride) {
  const int lane = threadIdx.x & 31;
  const __bf16* r = p + (lane & 15) * stride + (lane >> 4) * 8;
  union { v16bf v; u32x4 q[2]; } u;
  u.q[0] = *reinterpret_cast<const u32x4*>(r);
  u.q[1] = *reinterpret_cast<const u32x4*>(r + 16);
  return u.v;
}

__device__ __forceinline__ v16bf load_a_f32cvt(const float* p, int stride) {
  const int lane = threadIdx.x & 31;
  const float* r = p + (lane & 15) * stride + (lane >> 4) * 8;
  v16bf v;
#pragma unroll
  for (int i = 0; i < 8; ++i) v[i] = (__bf16)r[i];
#pragma unroll
  for (int i = 0; i < 8; ++i) v[8 + i] = (__bf16)r[16 + i];
  return v;
}

// ---- 16-lane row reductions (C-frag: one row spans lanes 0-15 / 16-31) ----
__device__ __forceinline__ float rowmax16(float v) {
#pragma unroll
  for (int m = 8; m >= 1; m >>= 1) v = fmaxf(v, __shfl_xor(v, m, 32));
  return v;
}
__device__ __forceinline__ float rowsum16(float v) {
#pragma unroll
  for (int m = 8; m >= 1; m >>= 1) v += __shfl_xor(v, m, 32);
  return v;
}

// ---- CDNA5 async copy: global -> LDS, tracked by ASYNCcnt -----------------
__device__ __forceinline__ void async_ld_b128(unsigned lds_off, const void* g) {
  asm volatile("global_load_async_to_lds_b128 %0, %1, off"
               :: "v"(lds_off), "v"(g) : "memory");
}

// ---------------------------------------------------------------------------
// Weight transpose + f32->bf16 convert: Wt[n][k] = W[k][n]
// ---------------------------------------------------------------------------
__global__ __launch_bounds__(256) void transpose_cvt(const float* __restrict__ W,
                                                     __bf16* __restrict__ Wt) {
  int idx = blockIdx.x * 256 + threadIdx.x;   // 1M elements
  int k = idx >> 10, n = idx & 1023;
  Wt[n * DIMN + k] = (__bf16)W[k * DIMN + n];
}

// ---------------------------------------------------------------------------
// WMMA GEMM: C[4096,1024] = A[4096,1024] @ Wt^T + bias
// MODE 0: A=f32 query  -> q  bf16 [B,H,L,D]
// MODE 1: A=f32 key    -> k  bf16 [B,H,S,D]
// MODE 2: A=f32 value  -> v^T bf16 [B,H,D,S]
// MODE 3: A=bf16 attn  -> f32 d_out [B*L, DIM]
// ---------------------------------------------------------------------------
template <int MODE>
__global__ __launch_bounds__(256) void gemm_wmma(const void* __restrict__ Ap,
                                                 const __bf16* __restrict__ Wt,
                                                 const float* __restrict__ bias,
                                                 void* __restrict__ outp) {
  const int wave = threadIdx.x >> 5;
  const int lane = threadIdx.x & 31;
  const int m0 = blockIdx.y * 128 + wave * 16;
  const int n0 = blockIdx.x * 64;

  v8f c[4] = {};
  for (int k0 = 0; k0 < DIMN; k0 += 32) {
    v16bf a;
    if (MODE == 3) a = load_a_bf16((const __bf16*)Ap + m0 * DIMN + k0, DIMN);
    else           a = load_a_f32cvt((const float*)Ap + m0 * DIMN + k0, DIMN);
#pragma unroll
    for (int j = 0; j < 4; ++j) {
      v16bf b = load_a_bf16(Wt + (n0 + j * 16) * DIMN + k0, DIMN);
      c[j] = __builtin_amdgcn_wmma_f32_16x16x32_bf16(false, a, false, b,
                                                     (short)0, c[j], false, false);
    }
  }

  const int hg = lane >> 4;
  const int nn = lane & 15;
#pragma unroll
  for (int j = 0; j < 4; ++j) {
#pragma unroll
    for (int r = 0; r < 8; ++r) {
      int row = m0 + r + 8 * hg;          // global M index = b*L + l
      int col = n0 + j * 16 + nn;         // global N index = h*64 + d
      float v = c[j][r] + bias[col];
      if (MODE == 3) {
        ((float*)outp)[row * DIMN + col] = v;
      } else {
        int b_ = row >> 11, l_ = row & 2047;
        int h_ = col >> 6,  d_ = col & 63;
        __bf16* o = (__bf16*)outp;
        if (MODE == 2) o[((b_ * NH + h_) * HD + d_) * SS + l_] = (__bf16)v;  // v^T
        else           o[((b_ * NH + h_) * LL + l_) * HD + d_] = (__bf16)v;  // q/k
      }
    }
  }
}

// ---------------------------------------------------------------------------
// Flash attention, 64-wide S chunks. 8 waves/block share one (b,h): K and V^T
// tiles are staged once per block into LDS via async DMA (double-buffered).
// ---------------------------------------------------------------------------
#define NCHUNK (SS / 64)   // 32

// stage 64x64 bf16 K tile (contiguous 8KB) : 2 async b128 per thread
__device__ __forceinline__ void stage_k(const __bf16* g, __bf16* dst) {
  const int tid = threadIdx.x;
  unsigned l0 = (unsigned)(uintptr_t)dst + tid * 16u;
  const char* g0 = (const char*)g + tid * 16;
  async_ld_b128(l0, g0);
  async_ld_b128(l0 + 4096u, g0 + 4096);
}

// stage 64(d) x 64(s) bf16 V^T tile (row stride S) : 2 async b128 per thread
__device__ __forceinline__ void stage_v(const __bf16* vt_g, int s0, __bf16* dst) {
  const int tid = threadIdx.x;
#pragma unroll
  for (int pass = 0; pass < 2; ++pass) {
    int c = tid + pass * 256;            // 16B chunk index, 8 chunks per row
    int row = c >> 3, cc = c & 7;
    unsigned l = (unsigned)(uintptr_t)dst + (unsigned)c * 16u;
    const char* gp = (const char*)(vt_g + (long)row * SS + s0) + cc * 16;
    async_ld_b128(l, gp);
  }
}

__global__ __launch_bounds__(256) void attn_wmma(const __bf16* __restrict__ qb,
                                                 const __bf16* __restrict__ kb,
                                                 const __bf16* __restrict__ vt,
                                                 const int* __restrict__ mask,
                                                 const float* __restrict__ bias,
                                                 __bf16* __restrict__ attn_out) {
  __shared__ __bf16 kt[2][64 * 64];     // 2 x 8KB
  __shared__ __bf16 vs[2][64 * 64];     // 2 x 8KB
  __shared__ __bf16 pt[8][16 * 64];     // 8 x 2KB
  const int wave = threadIdx.x >> 5;
  const int lane = threadIdx.x & 31;
  const int t  = blockIdx.x * 8 + wave;  // 4096 wave-tiles
  const int lt = t & 127;                // L/16; all 8 waves share bh
  const int bh = t >> 7;                 // b*16 + h
  const int b_ = bh >> 4;

  const long qoff = (long)bh * LL * HD + (long)lt * 16 * HD;
  const long koff = (long)bh * SS * HD;
  const long voff = (long)bh * HD * SS;
  const long boff = ((long)bh * LL + lt * 16) * SS;
  const long moff = ((long)b_ * LL + lt * 16) * SS;

  const v16bf aq0 = load_a_bf16(qb + qoff, HD);        // d = 0..31
  const v16bf aq1 = load_a_bf16(qb + qoff + 32, HD);   // d = 32..63

  v8f acc[4] = {};
  float row_m[8], row_s[8];
#pragma unroll
  for (int r = 0; r < 8; ++r) { row_m[r] = -1e30f; row_s[r] = 0.f; }

  const int hg = lane >> 4;
  const int nn = lane & 15;

  // prologue: DMA chunk 0 into buffer 0
  stage_k(kb + koff, kt[0]);
  stage_v(vt + voff, 0, vs[0]);

  for (int i = 0; i < NCHUNK; ++i) {
    const int cur = i & 1;
    const int s0  = i * 64;
    if (i + 1 < NCHUNK) {               // DMA next chunk while computing this one
      stage_k(kb + koff + (long)(s0 + 64) * HD, kt[cur ^ 1]);
      stage_v(vt + voff, s0 + 64, vs[cur ^ 1]);
      asm volatile("s_wait_asynccnt 4" ::: "memory");   // chunk i's 4 loads done
    } else {
      asm volatile("s_wait_asynccnt 0" ::: "memory");
    }
    __syncthreads();                    // all waves' portions landed in LDS

    // ---- scores: Q (16x64) . K^T (64x64) -> 8 WMMAs ----------------------
    v8f sc[4] = {};
#pragma unroll
    for (int j = 0; j < 4; ++j) {
      v16bf bk0 = load_a_bf16(&kt[cur][(j * 16) * 64], 64);
      v16bf bk1 = load_a_bf16(&kt[cur][(j * 16) * 64 + 32], 64);
      sc[j] = __builtin_amdgcn_wmma_f32_16x16x32_bf16(false, aq0, false, bk0,
                                                      (short)0, sc[j], false, false);
      sc[j] = __builtin_amdgcn_wmma_f32_16x16x32_bf16(false, aq1, false, bk1,
                                                      (short)0, sc[j], false, false);
    }

    // ---- bias + mask + online softmax (dominant HBM stream) --------------
    float alpha[8];
#pragma unroll
    for (int r = 0; r < 8; ++r) {
      const int lrow = r + 8 * hg;
      const float* bp = bias + boff + (long)lrow * SS + s0 + nn;
      const int*   mp = mask + moff + (long)lrow * SS + s0 + nn;
      float v0 = sc[0][r] * SCALE_F + bp[0];
      float v1 = sc[1][r] * SCALE_F + bp[16];
      float v2 = sc[2][r] * SCALE_F + bp[32];
      float v3 = sc[3][r] * SCALE_F + bp[48];
      v0 = (mp[0]  == 0) ? -1e30f : v0;
      v1 = (mp[16] == 0) ? -1e30f : v1;
      v2 = (mp[32] == 0) ? -1e30f : v2;
      v3 = (mp[48] == 0) ? -1e30f : v3;
      float mx = fmaxf(fmaxf(v0, v1), fmaxf(v2, v3));
      float nm = fmaxf(row_m[r], rowmax16(mx));
      float al = __expf(row_m[r] - nm);
      row_m[r] = nm;
      float p0 = __expf(v0 - nm);
      float p1 = __expf(v1 - nm);
      float p2 = __expf(v2 - nm);
      float p3 = __expf(v3 - nm);
      row_s[r] = row_s[r] * al + rowsum16((p0 + p1) + (p2 + p3));
      alpha[r] = al;
      __bf16* lp = &pt[wave][lrow * 64];
      lp[nn]      = (__bf16)p0;
      lp[16 + nn] = (__bf16)p1;
      lp[32 + nn] = (__bf16)p2;
      lp[48 + nn] = (__bf16)p3;
    }

#pragma unroll
    for (int jn = 0; jn < 4; ++jn)
#pragma unroll
      for (int r = 0; r < 8; ++r) acc[jn][r] *= alpha[r];

    // ---- P.V: P (16x64) from LDS (transpose), V^T tiles -> 8 WMMAs -------
    v16bf ap0 = load_a_bf16(&pt[wave][0], 64);
    v16bf ap1 = load_a_bf16(&pt[wave][32], 64);
#pragma unroll
    for (int jn = 0; jn < 4; ++jn) {
      v16bf bv0 = load_a_bf16(&vs[cur][(jn * 16) * 64], 64);
      v16bf bv1 = load_a_bf16(&vs[cur][(jn * 16) * 64 + 32], 64);
      acc[jn] = __builtin_amdgcn_wmma_f32_16x16x32_bf16(false, ap0, false, bv0,
                                                        (short)0, acc[jn], false, false);
      acc[jn] = __builtin_amdgcn_wmma_f32_16x16x32_bf16(false, ap1, false, bv1,
                                                        (short)0, acc[jn], false, false);
    }
    __syncthreads();   // everyone done reading before next DMA overwrites
  }

  // ---- epilogue: normalize, write bf16 [B*L, H*D] for the Wo GEMM --------
#pragma unroll
  for (int r = 0; r < 8; ++r) {
    const int lrow = lt * 16 + r + 8 * hg;
    const float inv = 1.0f / fmaxf(row_s[r], 1e-20f);
    const int h_ = bh & 15;
#pragma unroll
    for (int jn = 0; jn < 4; ++jn) {
      attn_out[(long)(b_ * LL + lrow) * DIMN + h_ * HD + jn * 16 + nn] =
          (__bf16)(acc[jn][r] * inv);
    }
  }
}

// ---------------------------------------------------------------------------
extern "C" void kernel_launch(void* const* d_in, const int* in_sizes, int n_in,
                              void* d_out, int out_size, void* d_ws, size_t ws_size,
                              hipStream_t stream) {
  const float* query = (const float*)d_in[0];
  const float* key_  = (const float*)d_in[1];
  const float* value = (const float*)d_in[2];
  const int*   mask  = (const int*)d_in[3];
  const float* bias  = (const float*)d_in[4];
  const float* Wq = (const float*)d_in[5];
  const float* bq = (const float*)d_in[6];
  const float* Wk = (const float*)d_in[7];
  const float* bk = (const float*)d_in[8];
  const float* Wv = (const float*)d_in[9];
  const float* bv = (const float*)d_in[10];
  const float* Wo = (const float*)d_in[11];
  const float* bo = (const float*)d_in[12];
  float* out = (float*)d_out;

  // workspace layout (bf16 elements): 4x transposed weights, q, k, v^T, attn
  __bf16* base = (__bf16*)d_ws;
  const size_t WSZ = (size_t)DIMN * DIMN;          // 1M
  const size_t ASZ = (size_t)BB * LL * DIMN;       // 4M
  __bf16* Wqt = base;
  __bf16* Wkt = base + WSZ;
  __bf16* Wvt = base + 2 * WSZ;
  __bf16* Wot = base + 3 * WSZ;
  __bf16* qbf = base + 4 * WSZ;
  __bf16* kbf = qbf + ASZ;
  __bf16* vtb = kbf + ASZ;
  __bf16* att = vtb + ASZ;

  dim3 tgrid(4096);                                // 1M / 256
  transpose_cvt<<<tgrid, 256, 0, stream>>>(Wq, Wqt);
  transpose_cvt<<<tgrid, 256, 0, stream>>>(Wk, Wkt);
  transpose_cvt<<<tgrid, 256, 0, stream>>>(Wv, Wvt);
  transpose_cvt<<<tgrid, 256, 0, stream>>>(Wo, Wot);

  dim3 ggrid(DIMN / 64, (BB * LL) / 128);          // (16, 32)
  gemm_wmma<0><<<ggrid, 256, 0, stream>>>((const void*)query, Wqt, bq, (void*)qbf);
  gemm_wmma<1><<<ggrid, 256, 0, stream>>>((const void*)key_,  Wkt, bk, (void*)kbf);
  gemm_wmma<2><<<ggrid, 256, 0, stream>>>((const void*)value, Wvt, bv, (void*)vtb);

  attn_wmma<<<512, 256, 0, stream>>>(qbf, kbf, vtb, mask, bias, att);

  gemm_wmma<3><<<ggrid, 256, 0, stream>>>((const void*)att, Wot, bo, (void*)out);
}
